// DGLLayerHomogeneous_41291815584407
// MI455X (gfx1250) — matrix operand order repaired and checked by reading, not compiled, not run on hardware
//
#include <hip/hip_runtime.h>
#include <stdint.h>

// ---------------------------------------------------------------------------
// Types for CDNA5 WMMA (wave32): A/B fragments = 16 bf16 per lane (8 VGPRs),
// C/D = 8 f32 per lane.
// ---------------------------------------------------------------------------
typedef __attribute__((ext_vector_type(16))) __bf16 v16bf;
typedef __attribute__((ext_vector_type(8)))  float  v8f;
typedef int v4i_async __attribute__((vector_size(16)));   // matches builtin param

static constexpr int NUM_RNA   = 16384;
static constexpr int NUM_DIS   = 8192;
static constexpr int NN        = NUM_RNA + NUM_DIS;   // 24576
static constexpr int E_EDGES   = 786432;
static constexpr int DIM       = 256;                  // common_dim
static constexpr int HEADS     = 4;
static constexpr int OUT_F     = 64;
static constexpr int OUT_UNITS = 256;

#define TN 4           // 16-col tiles per wave
#define SROW 80        // LDS bytes per staged B row: 64B data + 16B pad (bank-safe)

#if __has_builtin(__builtin_amdgcn_global_load_async_to_lds_b128)
#define ASYNC_COPY 1
#else
#define ASYNC_COPY 0
#endif

__device__ __forceinline__ uint16_t f2bf(float f) {
  union { float f; uint32_t u; } c; c.f = f;
  uint32_t r = c.u + 0x7FFFu + ((c.u >> 16) & 1u);   // round-to-nearest-even
  return (uint16_t)(r >> 16);
}

struct ABFrag {
  union { v16bf v; uint4 q[2]; };
};

// A fragment: lane m (m=lane&15) holds row m, K-chunk kb..kb+15
// (lane>=16 shifted by +16).  Two contiguous 16B loads.
__device__ __forceinline__ void load_a(ABFrag& f, const uint16_t* Aptr, int kb) {
  f.q[0] = *(const uint4*)(Aptr + kb);
  f.q[1] = *(const uint4*)(Aptr + kb + 8);
}

// B fragment from row-major Bt (N x K): lane n holds Bt[n, kb..kb+15]
// (lane>=16: kb+16..kb+31).  One 32B contiguous read per lane.
__device__ __forceinline__ void load_b4(ABFrag (&f)[TN], const uint16_t* Bt,
                                        int n0, int brow, int bkof, int K, int kb) {
#pragma unroll
  for (int t = 0; t < TN; ++t) {
    const uint16_t* Bp = Bt + (size_t)(n0 + t * 16 + brow) * K + kb + bkof;
    f[t].q[0] = *(const uint4*)(Bp);
    f[t].q[1] = *(const uint4*)(Bp + 8);
  }
}

__device__ __forceinline__ void mma4(v8f (&acc)[TN], const ABFrag& a,
                                     const ABFrag (&b)[TN]) {
#pragma unroll
  for (int t = 0; t < TN; ++t)
    acc[t] = __builtin_amdgcn_wmma_f32_16x16x32_bf16(
        false, a.v, false, b[t].v, (short)0, acc[t], false, false);
}

// ---------------------------------------------------------------------------
// Generic bf16 WMMA GEMM:  C[M,N] = A[M,K] * Bt[N,K]^T (+ bias[N]).
// Ping-pong register double buffering (K % 64 == 0) so VMEM overlaps WMMA.
// One wave -> 16x64 output tile; block (32,4) -> 64 rows.
// ---------------------------------------------------------------------------
__global__ __launch_bounds__(128)
void gemm_bf16_wmma(const uint16_t* __restrict__ A,
                    const uint16_t* __restrict__ Bt,
                    const float*    __restrict__ bias,
                    float*          __restrict__ C,
                    uint16_t*       __restrict__ Cbf,
                    int M, int N, int K)
{
  const int lane = threadIdx.x;
  const int row0 = (blockIdx.y * 4 + threadIdx.y) * 16;
  const int n0   = blockIdx.x * (16 * TN);
  if (row0 >= M) return;

  const int arow = lane & 15;
  const int akof = (lane >> 4) * 16;
  const int brow = lane & 15;
  const int bkof = (lane >> 4) * 16;
  const uint16_t* Aptr = A + (size_t)(row0 + arow) * K + akof;

  v8f acc[TN] = {};
  ABFrag a[2];
  ABFrag b[2][TN];
  load_a(a[0], Aptr, 0);
  load_b4(b[0], Bt, n0, brow, bkof, K, 0);

  for (int kb = 0; kb < K; kb += 64) {
    load_a(a[1], Aptr, kb + 32);
    load_b4(b[1], Bt, n0, brow, bkof, K, kb + 32);
    mma4(acc, a[0], b[0]);
    if (kb + 64 < K) {
      load_a(a[0], Aptr, kb + 64);
      load_b4(b[0], Bt, n0, brow, bkof, K, kb + 64);
    }
    mma4(acc, a[1], b[1]);
  }

  const int crow  = row0 + (lane >> 4) * 8;
  const int ccol0 = lane & 15;
#pragma unroll
  for (int t = 0; t < TN; ++t) {
    const int col = n0 + t * 16 + ccol0;
    const float bv = bias ? bias[col] : 0.0f;
#pragma unroll
    for (int r = 0; r < 8; ++r) {
      const float v = acc[t][r] + bv;
      const size_t idx = (size_t)(crow + r) * N + col;
      C[idx] = v;
      if (Cbf) Cbf[idx] = f2bf(v);
    }
  }
}

// ---------------------------------------------------------------------------
// Score GEMM with LDS-staged B tiles (shared by all 4 waves of the block),
// double-buffered via CDNA5 async global->LDS copies when available.
// ---------------------------------------------------------------------------
__device__ __forceinline__ void stage_chunk16(const uint16_t* g, void* l) {
#if ASYNC_COPY
  __builtin_amdgcn_global_load_async_to_lds_b128(
      (__attribute__((address_space(1))) v4i_async*)(uintptr_t)g,
      (__attribute__((address_space(3))) v4i_async*)(uint32_t)(uintptr_t)l,
      0, 0);
#else
  *(uint4*)l = *(const uint4*)g;
#endif
}

__device__ __forceinline__ void lds_sync() {
#if ASYNC_COPY
#if __has_builtin(__builtin_amdgcn_s_wait_asynccnt)
  __builtin_amdgcn_s_wait_asynccnt(0);
#else
  asm volatile("s_wait_asynccnt 0x0" ::: "memory");
#endif
#endif
  __syncthreads();
}

__global__ __launch_bounds__(128)
void score_gemm_lds(const uint16_t* __restrict__ A,   // u_bf  (M x K)
                    const uint16_t* __restrict__ Bt,  // i_bf  (N x K)
                    float* __restrict__ C, int M, int N, int K)
{
  __shared__ __align__(16) uint8_t ldsB[2][64 * SROW];

  const int lane = threadIdx.x;
  const int tid  = threadIdx.y * 32 + lane;           // 0..127
  const int row0 = (blockIdx.y * 4 + threadIdx.y) * 16;
  const int n0   = blockIdx.x * 64;

  const int arow = lane & 15;
  const int akof = (lane >> 4) * 16;
  const uint16_t* Aptr = A + (size_t)(row0 + arow) * K + akof;

  // Stage 64 rows x 32 bf16 (4KB) of Bt into LDS buffer `buf`.
  // 256 16-byte chunks; 128 threads x 2 chunks.
  auto stageB = [&](int buf, int kb) {
#pragma unroll
    for (int j = 0; j < 2; ++j) {
      const int c = tid + j * 128;
      const int r = c >> 2, sub = c & 3;
      const uint16_t* g = Bt + (size_t)(n0 + r) * K + kb + sub * 8;
      stage_chunk16(g, &ldsB[buf][r * SROW + sub * 16]);
    }
  };

  auto mma_lds = [&](v8f (&acc)[TN], const ABFrag& a, int buf) {
    const int c0 = (lane >> 4) * 2;                   // 16B chunk within row
#pragma unroll
    for (int t = 0; t < TN; ++t) {
      const int r = t * 16 + (lane & 15);
      ABFrag b;
      const uint4* lrow = (const uint4*)&ldsB[buf][r * SROW];
      b.q[0] = lrow[c0];
      b.q[1] = lrow[c0 + 1];
      acc[t] = __builtin_amdgcn_wmma_f32_16x16x32_bf16(
          false, a.v, false, b.v, (short)0, acc[t], false, false);
    }
  };

  v8f acc[TN] = {};
  ABFrag a[2];
  stageB(0, 0);
  load_a(a[0], Aptr, 0);
  lds_sync();                                         // buf0 ready

  for (int kb = 0; kb < K; kb += 64) {                // K % 64 == 0
    stageB(1, kb + 32);                               // fill buf1
    load_a(a[1], Aptr, kb + 32);
    mma_lds(acc, a[0], 0);                            // consume buf0
    lds_sync();                                       // buf1 ready, buf0 free
    if (kb + 64 < K) {
      stageB(0, kb + 64);                             // refill buf0
      load_a(a[0], Aptr, kb + 64);
    }
    mma_lds(acc, a[1], 1);                            // consume buf1
    if (kb + 64 < K) lds_sync();                      // buf0 ready, buf1 free
  }

  const int crow  = row0 + (lane >> 4) * 8;
  const int ccol0 = lane & 15;
#pragma unroll
  for (int t = 0; t < TN; ++t) {
    const int col = n0 + t * 16 + ccol0;
#pragma unroll
    for (int r = 0; r < 8; ++r)
      C[(size_t)(crow + r) * N + col] = acc[t][r];
  }
}

// ---------------------------------------------------------------------------
// Prep kernels.
// ---------------------------------------------------------------------------
__global__ void cvt_bf16_kernel(const float* __restrict__ in,
                                uint16_t* __restrict__ out, int n) {
  int i = blockIdx.x * blockDim.x + threadIdx.x;
  if (i < n) out[i] = f2bf(in[i]);
}

__global__ void transpose256_bf_kernel(const float* __restrict__ W,
                                       uint16_t* __restrict__ Wt) {
  int k = blockIdx.y * 16 + threadIdx.y;
  int n = blockIdx.x * 16 + threadIdx.x;
  Wt[(size_t)n * 256 + k] = f2bf(W[(size_t)k * 256 + n]);
}

__global__ void fill_kernel(float* __restrict__ p, float v, int n) {
  int i = blockIdx.x * blockDim.x + threadIdx.x;
  if (i < n) p[i] = v;
}

// ---------------------------------------------------------------------------
// el/er: one wave per (node, head), wave32 shuffle reduction.
// ---------------------------------------------------------------------------
__global__ __launch_bounds__(128)
void el_er_kernel(const float* __restrict__ h,
                  const float* __restrict__ attn_l,
                  const float* __restrict__ attn_r,
                  float* __restrict__ el, float* __restrict__ er) {
  const int node = blockIdx.x;
  const int hd   = threadIdx.y;
  const int lane = threadIdx.x;
  const float* row = h + (size_t)node * DIM + hd * OUT_F;
  const float a0 = row[lane], a1 = row[lane + 32];
  float l = a0 * attn_l[hd * OUT_F + lane] + a1 * attn_l[hd * OUT_F + lane + 32];
  float r = a0 * attn_r[hd * OUT_F + lane] + a1 * attn_r[hd * OUT_F + lane + 32];
#pragma unroll
  for (int off = 16; off > 0; off >>= 1) {
    l += __shfl_xor(l, off, 32);
    r += __shfl_xor(r, off, 32);
  }
  if (lane == 0) { el[node * HEADS + hd] = l; er[node * HEADS + hd] = r; }
}

__device__ __forceinline__ float leaky(float v) {
  return v > 0.0f ? v : 0.2f * v;
}

__global__ void edge_max_kernel(const int* __restrict__ src,
                                const int* __restrict__ dst,
                                const float* __restrict__ el,
                                const float* __restrict__ er,
                                float* __restrict__ maxb, int E) {
  int e = blockIdx.x * blockDim.x + threadIdx.x;
  if (e >= E) return;
  const int s = src[e], d = dst[e];
#pragma unroll
  for (int hd = 0; hd < HEADS; ++hd) {
    float v = leaky(el[s * HEADS + hd] + er[d * HEADS + hd]);
    atomicMax(&maxb[d * HEADS + hd], v);
  }
}

__global__ void edge_exp_kernel(const int* __restrict__ src,
                                const int* __restrict__ dst,
                                const float* __restrict__ el,
                                const float* __restrict__ er,
                                const float* __restrict__ maxb,
                                float* __restrict__ denom,
                                float* __restrict__ wbuf, int E) {
  int e = blockIdx.x * blockDim.x + threadIdx.x;
  if (e >= E) return;
  const int s = src[e], d = dst[e];
#pragma unroll
  for (int hd = 0; hd < HEADS; ++hd) {
    float v = leaky(el[s * HEADS + hd] + er[d * HEADS + hd]);
    float w = __expf(v - maxb[d * HEADS + hd]);
    wbuf[(size_t)e * HEADS + hd] = w;
    atomicAdd(&denom[d * HEADS + hd], w);
  }
}

__global__ __launch_bounds__(256)
void edge_agg_kernel(const int* __restrict__ src,
                     const int* __restrict__ dst,
                     const float* __restrict__ h,
                     const float* __restrict__ wbuf,
                     const float* __restrict__ denom,
                     float* __restrict__ rst) {
  const int e  = blockIdx.x;
  const int hd = threadIdx.x >> 6;
  const int o  = threadIdx.x & 63;
  const int s = src[e], d = dst[e];
  const float alpha = wbuf[(size_t)e * HEADS + hd] / denom[d * HEADS + hd];
  const int c = hd * OUT_F + o;
  atomicAdd(&rst[(size_t)d * DIM + c], alpha * h[(size_t)s * DIM + c]);
}

// relu(elu(x + bias)) == relu(x + bias): fuse bias + relu + bf16 cast.
__global__ void act_kernel(const float* __restrict__ rst,
                           const float* __restrict__ bias_gat,
                           uint16_t* __restrict__ act_bf, int n) {
  int i = blockIdx.x * blockDim.x + threadIdx.x;
  if (i >= n) return;
  float v = rst[i] + bias_gat[i & (DIM - 1)];
  act_bf[i] = f2bf(v > 0.0f ? v : 0.0f);
}

// ---------------------------------------------------------------------------
// Host-side orchestration.
// ---------------------------------------------------------------------------
extern "C" void kernel_launch(void* const* d_in, const int* in_sizes, int n_in,
                              void* d_out, int out_size, void* d_ws, size_t ws_size,
                              hipStream_t stream) {
  const float* ufeat    = (const float*)d_in[0];
  const float* ifeat    = (const float*)d_in[1];
  const int*   src      = (const int*)d_in[2];
  const int*   dst      = (const int*)d_in[3];
  const float* W_gat    = (const float*)d_in[4];
  const float* attn_l   = (const float*)d_in[5];
  const float* attn_r   = (const float*)d_in[6];
  const float* bias_gat = (const float*)d_in[7];
  const float* W_u      = (const float*)d_in[8];
  const float* b_u      = (const float*)d_in[9];
  const float* W_i      = (const float*)d_in[10];
  const float* b_i      = (const float*)d_in[11];

  // d_out layout: score (16384*8192) | u (16384*256) | i (8192*256), all f32
  float* score = (float*)d_out;
  float* u_out = score + (size_t)NUM_RNA * NUM_DIS;
  float* i_out = u_out + (size_t)NUM_RNA * OUT_UNITS;

  char* ws = (char*)d_ws;
  auto carve = [&](size_t bytes) -> char* {
    char* p = ws;
    ws += (bytes + 255) & ~(size_t)255;
    return p;
  };
  uint16_t* feat_bf = (uint16_t*)carve((size_t)NN * DIM * 2);
  uint16_t* Wt_gat  = (uint16_t*)carve((size_t)DIM * DIM * 2);
  uint16_t* Wt_u    = (uint16_t*)carve((size_t)DIM * DIM * 2);
  uint16_t* Wt_i    = (uint16_t*)carve((size_t)DIM * DIM * 2);
  float*    h       = (float*)   carve((size_t)NN * DIM * 4);
  float*    el      = (float*)   carve((size_t)NN * HEADS * 4);
  float*    er      = (float*)   carve((size_t)NN * HEADS * 4);
  float*    maxb    = (float*)   carve((size_t)NN * HEADS * 4);
  float*    denom   = (float*)   carve((size_t)NN * HEADS * 4);
  float*    wbuf    = (float*)   carve((size_t)E_EDGES * HEADS * 4);
  float*    rst     = (float*)   carve((size_t)NN * DIM * 4);
  uint16_t* act_bf  = (uint16_t*)carve((size_t)NN * DIM * 2);
  uint16_t* u_bf    = (uint16_t*)carve((size_t)NUM_RNA * OUT_UNITS * 2);
  uint16_t* i_bf    = (uint16_t*)carve((size_t)NUM_DIS * OUT_UNITS * 2);

  const dim3 gemm_blk(32, 4);

  // ---- prep -----------------------------------------------------------------
  cvt_bf16_kernel<<<(NUM_RNA * DIM + 255) / 256, 256, 0, stream>>>(
      ufeat, feat_bf, NUM_RNA * DIM);
  cvt_bf16_kernel<<<(NUM_DIS * DIM + 255) / 256, 256, 0, stream>>>(
      ifeat, feat_bf + (size_t)NUM_RNA * DIM, NUM_DIS * DIM);
  transpose256_bf_kernel<<<dim3(16, 16), dim3(16, 16), 0, stream>>>(W_gat, Wt_gat);
  transpose256_bf_kernel<<<dim3(16, 16), dim3(16, 16), 0, stream>>>(W_u, Wt_u);
  transpose256_bf_kernel<<<dim3(16, 16), dim3(16, 16), 0, stream>>>(W_i, Wt_i);

  // ---- h = feat @ W_gat ------------------------------------------------------
  gemm_bf16_wmma<<<dim3(DIM / 64, NN / 64), gemm_blk, 0, stream>>>(
      feat_bf, Wt_gat, nullptr, h, nullptr, NN, DIM, DIM);

  // ---- attention logits ------------------------------------------------------
  el_er_kernel<<<NN, dim3(32, 4), 0, stream>>>(h, attn_l, attn_r, el, er);

  // ---- init segment buffers --------------------------------------------------
  fill_kernel<<<(NN * HEADS + 255) / 256, 256, 0, stream>>>(maxb, -3.0e38f,
                                                            NN * HEADS);
  (void)hipMemsetAsync(denom, 0, (size_t)NN * HEADS * 4, stream);
  (void)hipMemsetAsync(rst, 0, (size_t)NN * DIM * 4, stream);

  // ---- edge softmax + aggregation --------------------------------------------
  edge_max_kernel<<<E_EDGES / 256, 256, 0, stream>>>(src, dst, el, er, maxb,
                                                     E_EDGES);
  edge_exp_kernel<<<E_EDGES / 256, 256, 0, stream>>>(src, dst, el, er, maxb,
                                                     denom, wbuf, E_EDGES);
  edge_agg_kernel<<<E_EDGES, 256, 0, stream>>>(src, dst, h, wbuf, denom, rst);

  // ---- bias + relu + bf16 ----------------------------------------------------
  act_kernel<<<(NN * DIM + 255) / 256, 256, 0, stream>>>(rst, bias_gat, act_bf,
                                                         NN * DIM);

  // ---- u / i projections -----------------------------------------------------
  gemm_bf16_wmma<<<dim3(OUT_UNITS / 64, NUM_RNA / 64), gemm_blk, 0, stream>>>(
      act_bf, Wt_u, b_u, u_out, u_bf, NUM_RNA, OUT_UNITS, DIM);
  gemm_bf16_wmma<<<dim3(OUT_UNITS / 64, NUM_DIS / 64), gemm_blk, 0, stream>>>(
      act_bf + (size_t)NUM_RNA * DIM, Wt_i, b_i, i_out, i_bf, NUM_DIS,
      OUT_UNITS, DIM);

  // ---- score = u @ i^T (dominant GEMM, LDS/async-staged B) -------------------
  score_gemm_lds<<<dim3(NUM_DIS / 64, NUM_RNA / 64), gemm_blk, 0, stream>>>(
      u_bf, i_bf, score, NUM_RNA, NUM_DIS, OUT_UNITS);
}
